// GNNModel_39298950758659
// MI455X (gfx1250) — matrix operand order repaired and checked by reading, not compiled, not run on hardware
//
#include <hip/hip_runtime.h>
#include <hip/hip_bf16.h>

typedef __attribute__((ext_vector_type(16))) __bf16 v16bf;
typedef __attribute__((ext_vector_type(8)))  float  v8f;

#define BN_EPS 1e-5f

// ---------------------------------------------------------------------------
// Degree / normalization
// ---------------------------------------------------------------------------
__global__ void gcn_deg_init(float* __restrict__ deg, int n) {
    int i = blockIdx.x * blockDim.x + threadIdx.x;
    if (i < n) deg[i] = 1.0f;                  // self-loop weight 1
}

__global__ void gcn_deg_accum(const long long* __restrict__ ei,
                              const float* __restrict__ ew,
                              float* __restrict__ deg, int E) {
    int e = blockIdx.x * blockDim.x + threadIdx.x;
    if (e < E) atomicAdd(&deg[(size_t)ei[E + e]], ew[e]);   // dst row of edge_index
}

__global__ void gcn_deg_rsqrt(float* __restrict__ deg, int n) {
    int i = blockIdx.x * blockDim.x + threadIdx.x;
    if (i < n) deg[i] = rsqrtf(deg[i]);        // deg >= 1 always
}

// ---------------------------------------------------------------------------
// Pack W[128 x HOUT] (f32, row-major) into bf16 WMMA B-fragments:
// Wp[((ct*4 + kk)*32 + lane)*16 + t]  holds  W[k, ct*16 + (lane&15)]
// with k = kk*32 + (lane>>4)*16 + t   (contiguous K order, same as A, 16 consecutive
// K per lane, lane half selects K sub-range, lane&15 selects column).
// ---------------------------------------------------------------------------
template <int HOUT>
__global__ void gcn_pack_w(const float* __restrict__ W, __bf16* __restrict__ Wp) {
    int idx = blockIdx.x * blockDim.x + threadIdx.x;
    if (idx >= HOUT * 128) return;
    int t    = idx & 15;
    int lane = (idx >> 4) & 31;
    int kk   = (idx >> 9) & 3;
    int ct   = idx >> 11;
    int k    = kk * 32 + (lane >> 4) * 16 + t;
    int col  = ct * 16 + (lane & 15);
    Wp[idx] = (__bf16)W[(size_t)k * HOUT + col];
}

// ---------------------------------------------------------------------------
// GEMM: Out[N x HOUT] = A[N x 128] @ W[128 x HOUT] via v_wmma_f32_16x16x32_bf16
// One wave owns a 16-row stripe: A fragments loaded/converted ONCE into
// registers (4 x v16bf), then reused across all HOUT/16 column tiles with
// pre-packed contiguous bf16 B fragments (2 x b128 loads per K-step).
// ---------------------------------------------------------------------------
template <int HOUT>
__global__ __launch_bounds__(128)
void gcn_gemm_wmma(const float* __restrict__ A, const __bf16* __restrict__ Wp,
                   float* __restrict__ Out, int nrows) {
    const int lane = threadIdx.x & 31;
    const int wave = threadIdx.x >> 5;
    const int m    = lane & 15;        // A row / D col within tile
    const int kh   = lane >> 4;        // lane half selects K sub-range
    const int rowTile = blockIdx.x * 4 + wave;       // wave-uniform
    if (rowTile * 16 >= nrows) return;               // wave-uniform exit

    const int row = rowTile * 16 + m;

    // ---- A fragments for all 4 K-steps, converted to bf16 once ----
    // lanes 0-15 hold K=[koff..koff+15];
    // lanes 16-31 hold K=[koff+16..koff+31]
    v16bf afrag[4];
    #pragma unroll
    for (int kk = 0; kk < 4; ++kk) {
        const float* ap = A + (size_t)row * 128 + kk * 32 + kh * 16;
        const float4 f0 = *(const float4*)(ap + 0);
        const float4 f1 = *(const float4*)(ap + 4);
        const float4 f2 = *(const float4*)(ap + 8);
        const float4 f3 = *(const float4*)(ap + 12);
        v16bf a;
        a[0]  = (__bf16)f0.x; a[1]  = (__bf16)f0.y; a[2]  = (__bf16)f0.z; a[3]  = (__bf16)f0.w;
        a[4]  = (__bf16)f1.x; a[5]  = (__bf16)f1.y; a[6]  = (__bf16)f1.z; a[7]  = (__bf16)f1.w;
        a[8]  = (__bf16)f2.x; a[9]  = (__bf16)f2.y; a[10] = (__bf16)f2.z; a[11] = (__bf16)f2.w;
        a[12] = (__bf16)f3.x; a[13] = (__bf16)f3.y; a[14] = (__bf16)f3.z; a[15] = (__bf16)f3.w;
        afrag[kk] = a;
    }

    // ---- sweep column tiles, reusing A fragments ----
    #pragma unroll
    for (int ct = 0; ct < HOUT / 16; ++ct) {
        v8f acc = {};
        #pragma unroll
        for (int kk = 0; kk < 4; ++kk) {
            const v16bf b = *(const v16bf*)(Wp + (((size_t)ct * 4 + kk) * 32 + lane) * 16);
            acc = __builtin_amdgcn_wmma_f32_16x16x32_bf16(
                      false, afrag[kk], false, b, (short)0, acc, false, false);
        }
        // D layout: VGPR r -> (M = r + 8*kh, N = lane&15)
        #pragma unroll
        for (int r = 0; r < 8; ++r) {
            const int orow = rowTile * 16 + r + kh * 8;
            Out[(size_t)orow * HOUT + ct * 16 + m] = acc[r];
        }
    }
}

// ---------------------------------------------------------------------------
// agg[i, :] = xw[i, :] * dis[i]^2 + bias   (self-loop term + bias)
// ---------------------------------------------------------------------------
template <int H>
__global__ void gcn_selfloop_bias(const float* __restrict__ xw,
                                  const float* __restrict__ dis,
                                  const float* __restrict__ bias,
                                  float* __restrict__ agg, int n) {
    constexpr int PER = H / 4;
    int g = blockIdx.x * blockDim.x + threadIdx.x;
    int i = g / PER, c = g % PER;
    if (i >= n) return;
    float s = dis[i]; s = s * s;           // 1/deg
    float4 v = ((const float4*)(xw + (size_t)i * H))[c];
    float4 bb = ((const float4*)bias)[c];
    float4 o = { v.x * s + bb.x, v.y * s + bb.y, v.z * s + bb.z, v.w * s + bb.w };
    ((float4*)(agg + (size_t)i * H))[c] = o;
}

// ---------------------------------------------------------------------------
// Edge scatter: agg[dst, :] += xw[src, :] * (dis[src] * w * dis[dst])
// Wave-coalesced float4 gather of xw[src] row, per-lane f32 atomics into
// L2-resident agg rows (feature tables fit in the 192 MB L2).
// ---------------------------------------------------------------------------
template <int H>
__global__ void gcn_edge_scatter(const float* __restrict__ xw,
                                 const float* __restrict__ dis,
                                 const long long* __restrict__ ei,
                                 const float* __restrict__ ew,
                                 float* __restrict__ agg, int E) {
    constexpr int PER = H / 4;
    int g = blockIdx.x * blockDim.x + threadIdx.x;
    int e = g / PER, c = g % PER;
    if (e >= E) return;
    long long s = ei[e];
    long long d = ei[E + e];
    float coef = dis[s] * ew[e] * dis[d];
    float4 v = ((const float4*)(xw + (size_t)s * H))[c];
    float* dp = agg + (size_t)d * H + c * 4;
    atomicAdd(dp + 0, v.x * coef);
    atomicAdd(dp + 1, v.y * coef);
    atomicAdd(dp + 2, v.z * coef);
    atomicAdd(dp + 3, v.w * coef);
}

// ---------------------------------------------------------------------------
// BatchNorm over relu(h): stats -> finalize -> apply (in place)
// ---------------------------------------------------------------------------
__global__ void gcn_zero_stats(float* __restrict__ stats, int n) {
    int i = blockIdx.x * blockDim.x + threadIdx.x;
    if (i < n) stats[i] = 0.0f;
}

template <int H>
__global__ __launch_bounds__(256)
void gcn_bn_stats(const float* __restrict__ h, float* __restrict__ stats, int n) {
    constexpr int RPB = 256 / H;                        // rows per block-iter
    const int c = threadIdx.x & (H - 1);
    int row = blockIdx.x * RPB + (threadIdx.x / H);
    const int stride = gridDim.x * RPB;
    float s = 0.f, q = 0.f;
    for (; row < n; row += stride) {
        float v = h[(size_t)row * H + c];
        v = v > 0.f ? v : 0.f;                          // relu before BN stats
        s += v; q += v * v;
    }
    __shared__ float ls[256], lq[256];
    ls[threadIdx.x] = s; lq[threadIdx.x] = q;
    __syncthreads();
    if (threadIdx.x < H) {
        float ts = ls[threadIdx.x], tq = lq[threadIdx.x];
        #pragma unroll
        for (int r = 1; r < RPB; ++r) {
            ts += ls[threadIdx.x + r * H];
            tq += lq[threadIdx.x + r * H];
        }
        atomicAdd(&stats[c], ts);
        atomicAdd(&stats[H + c], tq);
    }
}

template <int H>
__global__ void gcn_bn_finalize(const float* __restrict__ stats,
                                const float* __restrict__ g,
                                const float* __restrict__ be,
                                float* __restrict__ sc, int n) {
    int c = threadIdx.x;
    if (c < H) {
        float inv_n = 1.0f / (float)n;
        float mean = stats[c] * inv_n;
        float var  = stats[H + c] * inv_n - mean * mean;
        float k    = g[c] * rsqrtf(var + BN_EPS);
        sc[c]      = k;
        sc[H + c]  = be[c] - mean * k;
    }
}

template <int H>
__global__ void gcn_bn_apply(float* __restrict__ h, const float* __restrict__ sc,
                             int n) {
    int idx = blockIdx.x * blockDim.x + threadIdx.x;
    if (idx >= n * H) return;
    int c = idx & (H - 1);
    float v = h[idx];
    v = v > 0.f ? v : 0.f;
    h[idx] = v * sc[c] + sc[H + c];
}

// ---------------------------------------------------------------------------
// Orchestration
// ---------------------------------------------------------------------------
extern "C" void kernel_launch(void* const* d_in, const int* in_sizes, int n_in,
                              void* d_out, int out_size, void* d_ws, size_t ws_size,
                              hipStream_t stream) {
    const float*     x   = (const float*)d_in[0];
    const long long* ei  = (const long long*)d_in[1];   // [2, E] int64
    const float*     ew  = (const float*)d_in[2];
    const float*     W1  = (const float*)d_in[3];
    const float*     b1  = (const float*)d_in[4];
    const float*     g1  = (const float*)d_in[5];
    const float*     be1 = (const float*)d_in[6];
    const float*     W2  = (const float*)d_in[7];
    const float*     b2  = (const float*)d_in[8];
    const float*     g2  = (const float*)d_in[9];
    const float*     be2 = (const float*)d_in[10];
    const float*     W3  = (const float*)d_in[11];
    const float*     b3  = (const float*)d_in[12];

    const int N = in_sizes[0] / 128;
    const int E = in_sizes[2];

    float* ws    = (float*)d_ws;
    float* dis   = ws;                               // N
    float* bufX  = dis  + N;                         // N*128 (XW)
    float* bufA  = bufX + (size_t)N * 128;           // N*128 (agg / features)
    float* stats = bufA + (size_t)N * 128;           // 256
    float* sc    = stats + 256;                      // 256
    __bf16* w1p  = (__bf16*)(sc + 256);              // 128*128 bf16 packed
    __bf16* w2p  = w1p + 128 * 128;                  // 128*128
    __bf16* w3p  = w2p + 128 * 128;                  // 64*128
    float* out   = (float*)d_out;                    // N*64

    const int T = 256;

    // --- pack weights into WMMA B-fragment bf16 layout (tiny, once) ---
    gcn_pack_w<128><<<(128 * 128 + T - 1) / T, T, 0, stream>>>(W1, w1p);
    gcn_pack_w<128><<<(128 * 128 + T - 1) / T, T, 0, stream>>>(W2, w2p);
    gcn_pack_w<64> <<<(64 * 128 + T - 1) / T, T, 0, stream>>>(W3, w3p);

    // --- degree / normalization coefficients ---
    gcn_deg_init  <<<(N + T - 1) / T, T, 0, stream>>>(dis, N);
    gcn_deg_accum <<<(E + T - 1) / T, T, 0, stream>>>(ei, ew, dis, E);
    gcn_deg_rsqrt <<<(N + T - 1) / T, T, 0, stream>>>(dis, N);

    const int gemmGrid = (N / 16 + 3) / 4;           // 4 waves (row tiles) / block
    const int nh4  = (int)(((size_t)N * 128 / 4 + T - 1) / T);
    const int eh32 = (int)(((size_t)E * 32 + T - 1) / T);
    const int nh2  = (int)(((size_t)N * 64 / 4 + T - 1) / T);
    const int eh16 = (int)(((size_t)E * 16 + T - 1) / T);

    // ---------------- Layer 1 ----------------
    gcn_gemm_wmma<128><<<gemmGrid, 128, 0, stream>>>(x, w1p, bufX, N);
    gcn_selfloop_bias<128><<<nh4, T, 0, stream>>>(bufX, dis, b1, bufA, N);
    gcn_edge_scatter<128><<<eh32, T, 0, stream>>>(bufX, dis, ei, ew, bufA, E);
    gcn_zero_stats<<<1, 256, 0, stream>>>(stats, 256);
    gcn_bn_stats<128><<<512, 256, 0, stream>>>(bufA, stats, N);
    gcn_bn_finalize<128><<<1, 128, 0, stream>>>(stats, g1, be1, sc, N);
    gcn_bn_apply<128><<<nh4 * 4, T, 0, stream>>>(bufA, sc, N);

    // ---------------- Layer 2 ----------------
    gcn_gemm_wmma<128><<<gemmGrid, 128, 0, stream>>>(bufA, w2p, bufX, N);
    gcn_selfloop_bias<128><<<nh4, T, 0, stream>>>(bufX, dis, b2, bufA, N);
    gcn_edge_scatter<128><<<eh32, T, 0, stream>>>(bufX, dis, ei, ew, bufA, E);
    gcn_zero_stats<<<1, 256, 0, stream>>>(stats, 256);
    gcn_bn_stats<128><<<512, 256, 0, stream>>>(bufA, stats, N);
    gcn_bn_finalize<128><<<1, 128, 0, stream>>>(stats, g2, be2, sc, N);
    gcn_bn_apply<128><<<nh4 * 4, T, 0, stream>>>(bufA, sc, N);

    // ---------------- Layer 3 (output, H=64, no BN) ----------------
    gcn_gemm_wmma<64><<<gemmGrid, 128, 0, stream>>>(bufA, w3p, bufX, N);
    gcn_selfloop_bias<64><<<nh2, T, 0, stream>>>(bufX, dis, b3, out, N);
    gcn_edge_scatter<64><<<eh16, T, 0, stream>>>(bufX, dis, ei, ew, out, E);
}
